// SparseAttention_25941602468385
// MI455X (gfx1250) — matrix operand order repaired and checked by reading, not compiled, not run on hardware
//
#include <hip/hip_runtime.h>
#include <hip/hip_bf16.h>

// ---------------------------------------------------------------------------
// Sparse attention for MI455X (gfx1250, wave32, WMMA).
//   scores = Q @ M^T  (8.6 GFLOP, compute bound)  -> bf16x3-split WMMA
//   top-k  -> per-query histogram threshold (LDS atomics); near-threshold
//             softmax weights are ~e^-29 of the max, so output is insensitive
//   output = softmax(sel scores) @ M_sel          -> also WMMA
// Memory tiles are staged in LDS pre-split into bf16 hi/lo so the inner loop
// does wide ds_load_b128 fragment reads with no per-fragment conversion VALU.
// Per-query score max is tracked in registers (v_max_f32) and reduced to LDS
// once at the end instead of per-tile atomics.
// ---------------------------------------------------------------------------

typedef __bf16 bf16_t;
typedef __attribute__((ext_vector_type(16))) __bf16 v16bf;
typedef __attribute__((ext_vector_type(8)))  __bf16 v8bf;
typedef __attribute__((ext_vector_type(4)))  __bf16 v4bf;
typedef __attribute__((ext_vector_type(8)))  float  v8f;

#define DDIM    128
#define NMEM    16384
#define QTOT    2048          // B*Q = 4*512
#define TOPK    512           // NMEM * 0.03125
#define QTILE   16            // queries per workgroup (one WMMA M tile)
#define NTILE   128           // memory rows staged per iteration
#define NWAVES  8
#define BLOCK   (NWAVES * 32)
#define NBINS   1024
#define BIN_LO  (-64.0f)
#define BIN_SCALE 8.0f        // bin width 0.125; scores ~ N(0, 11.3)

// ----- order-preserving float <-> uint key (for the final atomicMax only) --
__device__ __forceinline__ unsigned f2key(float f) {
  unsigned u = __float_as_uint(f);
  return (u & 0x80000000u) ? ~u : (u | 0x80000000u);
}
__device__ __forceinline__ float key2f(unsigned k) {
  unsigned u = (k & 0x80000000u) ? (k ^ 0x80000000u) : ~k;
  return __uint_as_float(u);
}

// ----- f32 -> bf16 hi/lo split (3-term product keeps ~16 mantissa bits) ----
__device__ __forceinline__ bf16_t hi_of(float x) { return (bf16_t)x; }
__device__ __forceinline__ bf16_t lo_of(float x, bf16_t h) {
  return (bf16_t)(x - (float)h);
}

// A-matrix 16x32 bf16 K-index (ISA 7.12.2): lanes>=16 -> K+8, elems 8..15 -> K+16
__device__ __forceinline__ int kmapA(int lane, int e) {
  return ((lane >= 16) ? 8 : 0) + e + ((e >= 8) ? 8 : 0);
}

// Query A fragments from global f32 (done once per kernel; off the hot loop).
__device__ __forceinline__ void load_qfrag(const float* __restrict__ p,
                                           int lane, v16bf hi[4], v16bf lo[4]) {
  const int m = lane & 15;
#pragma unroll
  for (int c = 0; c < 4; ++c) {
    v16bf h, l;
#pragma unroll
    for (int e = 0; e < 16; ++e) {
      const float  x  = p[m * DDIM + 32 * c + kmapA(lane, e)];
      const bf16_t bh = hi_of(x);
      h[e] = bh;
      l[e] = lo_of(x, bh);
    }
    hi[c] = h;
    lo[c] = l;
  }
}

// B fragment (same K-map as A: lanes>=16 -> K+8, elems 8..15 -> K+16): two 16B LDS reads.
__device__ __forceinline__ v16bf load_bfrag(const bf16_t* base, int c, int koff) {
  const int b = 32 * c + (koff >> 1);
  v8bf b0 = *(const v8bf*)(base + b);
  v8bf b1 = *(const v8bf*)(base + b + 16);
  return __builtin_shufflevector(b0, b1, 0, 1, 2, 3, 4, 5, 6, 7,
                                 8, 9, 10, 11, 12, 13, 14, 15);
}

// A fragment from a pre-split bf16 row (A-map): two 16B LDS reads + shuffle.
__device__ __forceinline__ v16bf load_afrag(const bf16_t* row, int c, int lane) {
  const int b = 32 * c + ((lane >= 16) ? 8 : 0);
  v8bf a0 = *(const v8bf*)(row + b);
  v8bf a1 = *(const v8bf*)(row + b + 16);
  return __builtin_shufflevector(a0, a1, 0, 1, 2, 3, 4, 5, 6, 7,
                                 8, 9, 10, 11, 12, 13, 14, 15);
}

// acc += Ahi*Bhi + Ahi*Blo + Alo*Bhi over K=128 (12 x v_wmma_f32_16x16x32_bf16)
__device__ __forceinline__ v8f mma3(const v16bf ah[4], const v16bf al[4],
                                    const v16bf bh[4], const v16bf bl[4],
                                    v8f acc) {
#pragma unroll
  for (int c = 0; c < 4; ++c)
    acc = __builtin_amdgcn_wmma_f32_16x16x32_bf16(false, ah[c], false, bh[c],
                                                  (short)0, acc, false, false);
#pragma unroll
  for (int c = 0; c < 4; ++c)
    acc = __builtin_amdgcn_wmma_f32_16x16x32_bf16(false, ah[c], false, bl[c],
                                                  (short)0, acc, false, false);
#pragma unroll
  for (int c = 0; c < 4; ++c)
    acc = __builtin_amdgcn_wmma_f32_16x16x32_bf16(false, al[c], false, bh[c],
                                                  (short)0, acc, false, false);
  return acc;
}

// ---------------------------------------------------------------------------
// Kernel 1: per-query score max + approximate 512th-largest threshold.
// LDS: mh/ml 128x128 bf16 (64 KB) + hist 16x1024 u32 (64 KB).
// ---------------------------------------------------------------------------
__global__ void __launch_bounds__(BLOCK)
k_thresh(const float* __restrict__ q, const float* __restrict__ mem,
         float* __restrict__ thrmax /* [QTOT][2] = {thr, max} */) {
  extern __shared__ char smem[];
  bf16_t*   mh   = (bf16_t*)smem;              // [row][d] hi
  bf16_t*   ml   = mh + NTILE * DDIM;          // [row][d] lo
  unsigned* hist = (unsigned*)(ml + NTILE * DDIM);
  unsigned* maxk = hist + QTILE * NBINS;

  const int t = threadIdx.x, lane = t & 31, wv = t >> 5;
  const int wg = blockIdx.x;

  for (int i = t; i < QTILE * NBINS; i += BLOCK) hist[i] = 0u;
  if (t < QTILE) maxk[t] = 0u;
  __syncthreads();

  v16bf qhi[4], qlo[4];
  load_qfrag(q + (size_t)wg * QTILE * DDIM, lane, qhi, qlo);

  const int row  = wv * 16 + (lane & 15);
  const int koff = (lane >= 16) ? 16 : 0;

  float mx[8];  // per-lane running max for its 8 fixed queries (registers)
#pragma unroll
  for (int r = 0; r < 8; ++r) mx[r] = -3.4e38f;

  for (int tile = 0; tile < NMEM; tile += NTILE) {
    // stage 128x128 tile, converting f32 -> bf16 hi/lo once (packed cvt)
    const float4* src = (const float4*)(mem + (size_t)tile * DDIM);
    for (int i = t; i < NTILE * DDIM / 4; i += BLOCK) {
      const float4 v = src[i];
      const float  xs[4] = {v.x, v.y, v.z, v.w};
      v4bf h, l;
#pragma unroll
      for (int j = 0; j < 4; ++j) {
        const bf16_t bh = hi_of(xs[j]);
        const bf16_t bl = lo_of(xs[j], bh);
        h[j] = bh;
        l[j] = bl;
      }
      *(v4bf*)(mh + i * 4) = h;
      *(v4bf*)(ml + i * 4) = l;
    }
    if (tile + NTILE < NMEM)  // pull next tile toward L2 (global_prefetch_b8)
      __builtin_prefetch((const char*)(mem + (size_t)(tile + NTILE) * DDIM) +
                             t * 256, 0, 1);
    __syncthreads();

    // each wave: 16 queries x 16 memory rows; B frags are b128 LDS loads
    v16bf bh[4], bl[4];
#pragma unroll
    for (int c = 0; c < 4; ++c) {
      bh[c] = load_bfrag(mh + row * DDIM, c, koff);
      bl[c] = load_bfrag(ml + row * DDIM, c, koff);
    }
    v8f s = {0.f, 0.f, 0.f, 0.f, 0.f, 0.f, 0.f, 0.f};
    s = mma3(qhi, qlo, bh, bl, s);

    // C layout: elem r -> query m = r + 8*(lane>=16), row n = lane & 15
#pragma unroll
    for (int r = 0; r < 8; ++r) {
      const float sc = s[r];
      mx[r] = fmaxf(mx[r], sc);                 // register max, no atomics
      const int m = r + ((lane >= 16) ? 8 : 0);
      int bin = (int)((sc - BIN_LO) * BIN_SCALE);
      bin = bin < 0 ? 0 : (bin > NBINS - 1 ? NBINS - 1 : bin);
      atomicAdd(&hist[m * NBINS + bin], 1u);
    }
    __syncthreads();
  }

  // one max-key reduction per query at the end
#pragma unroll
  for (int r = 0; r < 8; ++r)
    atomicMax(&maxk[r + ((lane >= 16) ? 8 : 0)], f2key(mx[r]));
  __syncthreads();

  if (t < QTILE) {  // scan histogram from the top for the top-512 cutoff
    int cum = 0, b = NBINS - 1;
    for (; b > 0; --b) {
      cum += (int)hist[t * NBINS + b];
      if (cum >= TOPK) break;
    }
    const int gq = wg * QTILE + t;
    thrmax[gq * 2 + 0] = BIN_LO + (float)b * (1.0f / BIN_SCALE);
    thrmax[gq * 2 + 1] = key2f(maxk[t]);
  }
}

// ---------------------------------------------------------------------------
// Kernel 2: recompute scores, softmax over selected rows, weighted sum (WMMA).
// LDS: mh/ml row-major + mth/mtl transposed (128 KB) + split weight tile (8 KB).
// ---------------------------------------------------------------------------
__global__ void __launch_bounds__(BLOCK)
k_attend(const float* __restrict__ q, const float* __restrict__ mem,
         const float* __restrict__ thrmax, float* __restrict__ out) {
  extern __shared__ char smem[];
  bf16_t* mh  = (bf16_t*)smem;          // [row][d] hi
  bf16_t* ml  = mh  + NTILE * DDIM;     // [row][d] lo
  bf16_t* mth = ml  + NTILE * DDIM;     // [d][row] hi (for weighted-sum B)
  bf16_t* mtl = mth + NTILE * DDIM;     // [d][row] lo
  bf16_t* wh  = mtl + NTILE * DDIM;     // [q][row] weight hi
  bf16_t* wl  = wh  + QTILE * NTILE;    // [q][row] weight lo
  float*  zacc = (float*)(wl + QTILE * NTILE);

  const int t = threadIdx.x, lane = t & 31, wv = t >> 5;
  const int wg = blockIdx.x;
  if (t < QTILE) zacc[t] = 0.f;

  v16bf qhi[4], qlo[4];
  load_qfrag(q + (size_t)wg * QTILE * DDIM, lane, qhi, qlo);

  float thrv[8], maxv[8], zr[8];
#pragma unroll
  for (int r = 0; r < 8; ++r) {
    const int gq = wg * QTILE + r + ((lane >= 16) ? 8 : 0);
    thrv[r] = thrmax[gq * 2 + 0];
    maxv[r] = thrmax[gq * 2 + 1];
    zr[r]   = 0.f;
  }

  const int row  = wv * 16 + (lane & 15);  // this wave's score rows
  const int dcol = wv * 16 + (lane & 15);  // this wave's output D columns
  const int koff = (lane >= 16) ? 16 : 0;

  v8f acc = {0.f, 0.f, 0.f, 0.f, 0.f, 0.f, 0.f, 0.f};  // 16 q x 16 D cols

  for (int tile = 0; tile < NMEM; tile += NTILE) {
    // stage tile split hi/lo, both row-major and transposed
    const float4* src = (const float4*)(mem + (size_t)tile * DDIM);
    for (int i = t; i < NTILE * DDIM / 4; i += BLOCK) {
      const float4 v = src[i];
      const float  xs[4] = {v.x, v.y, v.z, v.w};
      v4bf h, l;
#pragma unroll
      for (int j = 0; j < 4; ++j) {
        const bf16_t bh = hi_of(xs[j]);
        const bf16_t bl = lo_of(xs[j], bh);
        h[j] = bh;
        l[j] = bl;
      }
      *(v4bf*)(mh + i * 4) = h;
      *(v4bf*)(ml + i * 4) = l;
      const int r0 = (i * 4) >> 7, d0 = (i * 4) & 127;
#pragma unroll
      for (int j = 0; j < 4; ++j) {
        mth[(d0 + j) * NTILE + r0] = h[j];
        mtl[(d0 + j) * NTILE + r0] = l[j];
      }
    }
    if (tile + NTILE < NMEM)
      __builtin_prefetch((const char*)(mem + (size_t)(tile + NTILE) * DDIM) +
                             t * 256, 0, 1);
    __syncthreads();

    // scores for this 128-row tile (identical arithmetic to kernel 1)
    {
      v16bf bh[4], bl[4];
#pragma unroll
      for (int c = 0; c < 4; ++c) {
        bh[c] = load_bfrag(mh + row * DDIM, c, koff);
        bl[c] = load_bfrag(ml + row * DDIM, c, koff);
      }
      v8f s = {0.f, 0.f, 0.f, 0.f, 0.f, 0.f, 0.f, 0.f};
      s = mma3(qhi, qlo, bh, bl, s);
#pragma unroll
      for (int r = 0; r < 8; ++r) {
        const float sc = s[r];
        const float w  = (sc >= thrv[r]) ? __expf(sc - maxv[r]) : 0.f;
        zr[r] += w;
        const int m = r + ((lane >= 16) ? 8 : 0);
        const bf16_t hwv = hi_of(w);
        const bf16_t lwv = lo_of(w, hwv);
        wh[m * NTILE + row] = hwv;
        wl[m * NTILE + row] = lwv;
      }
    }
    __syncthreads();

    // acc[q][d] += sum_rows w[q][row] * mem[row][d]
    // A = weight tile (16 x 128), B = transposed memory tile columns
    {
      v16bf ah[4], al[4], bh[4], bl[4];
      const int m = lane & 15;
#pragma unroll
      for (int c = 0; c < 4; ++c) {
        ah[c] = load_afrag(wh + m * NTILE, c, lane);
        al[c] = load_afrag(wl + m * NTILE, c, lane);
        bh[c] = load_bfrag(mth + dcol * NTILE, c, koff);
        bl[c] = load_bfrag(mtl + dcol * NTILE, c, koff);
      }
      acc = mma3(ah, al, bh, bl, acc);
    }
    __syncthreads();
  }

  // reduce softmax denominators and write normalized output
#pragma unroll
  for (int r = 0; r < 8; ++r)
    atomicAdd(&zacc[r + ((lane >= 16) ? 8 : 0)], zr[r]);
  __syncthreads();

#pragma unroll
  for (int r = 0; r < 8; ++r) {
    const int m = r + ((lane >= 16) ? 8 : 0);
    out[((size_t)(wg * QTILE + m)) * DDIM + dcol] = acc[r] / zacc[m];
  }
}

// ---------------------------------------------------------------------------
extern "C" void kernel_launch(void* const* d_in, const int* in_sizes, int n_in,
                              void* d_out, int out_size, void* d_ws,
                              size_t ws_size, hipStream_t stream) {
  const float* q   = (const float*)d_in[0];   // [4,512,128] f32
  const float* mem = (const float*)d_in[1];   // [16384,128] f32
  float* out    = (float*)d_out;              // [4,512,128] f32
  float* thrmax = (float*)d_ws;               // [2048][2] f32 (16 KB)

  const int nwg = QTOT / QTILE;  // 128 workgroups
  const size_t smem1 = (size_t)NTILE * DDIM * 2 * 2 +      // mh+ml bf16
                       (size_t)QTILE * NBINS * 4 +         // hist
                       QTILE * 4;                          // maxk  (~128 KB)
  const size_t smem2 = (size_t)NTILE * DDIM * 2 * 4 +      // mh/ml/mth/mtl
                       (size_t)QTILE * NTILE * 2 * 2 +     // wh/wl
                       QTILE * 4;                          // zacc  (~136 KB)

  k_thresh<<<nwg, BLOCK, smem1, stream>>>(q, mem, thrmax);
  k_attend<<<nwg, BLOCK, smem2, stream>>>(q, mem, thrmax, out);
}
